// LoanTrajectoryModel_29978871726110
// MI455X (gfx1250) — matrix operand order repaired and checked by reading, not compiled, not run on hardware
//
#include <hip/hip_runtime.h>
#include <math.h>

// ---------------------------------------------------------------------------
// LoanTrajectoryModel for MI455X (gfx1250, wave32, WMMA).
// All large GEMMs go through v_wmma_f32_16x16x32_f16 (f16 in, f32 accumulate).
// Dominant cost: 100 diffusion steps x (30720x128x128) GEMM + 4 transformer
// layers; everything is L2-resident (~16MB per activation tensor).
// ---------------------------------------------------------------------------

typedef __attribute__((ext_vector_type(16))) _Float16 v16h;
typedef __attribute__((ext_vector_type(8)))  _Float16 v8h;
typedef __attribute__((ext_vector_type(8)))  float    v8f;

#define BCNT 256
#define SEQ  120
#define DIM  128
#define NHEAD 8
#define HDIM 16
#define NLAYER 4
#define DFFN 512
#define NMAC 9
#define NST  7
#define TSTEP 100
#define NROW (BCNT * SEQ)   // 30720

// ---------------------------------------------------------------------------
// Generic WMMA GEMM:  C[M,N] = act( A[M,K] @ B[K,N] + bias[N] )
// fp32 in memory; tiles converted to f16 in-register while staging to LDS,
// multiplied with v_wmma_f32_16x16x32_f16 (f32 accumulate).
// Block: 256 threads = 8 waves, wave grid 4(M) x 2(N); each wave owns a
// 16x32 output slice (two v8f accumulators) -> 64x64 block tile, K-step 32.
// Per K-step per wave: 2 WMMA, A fragment reused across both B fragments.
// Requires: M % 64 == 0, N % 64 == 0, K % 32 == 0 (true for every call here).
// act: 0 = none, 1 = relu, 2 = silu
// ---------------------------------------------------------------------------
__global__ __launch_bounds__(256) void wmma_gemm_kernel(
    const float* __restrict__ A, const float* __restrict__ B,
    const float* __restrict__ bias, float* __restrict__ C,
    int M, int N, int K, int act) {
  // LDS pitch 40 halves = 80B: keeps every 8-half chunk 16B-aligned for b128.
  __shared__ __align__(16) _Float16 As[64 * 40];   // [row m][k]
  __shared__ __align__(16) _Float16 Bs[64 * 40];   // [col n][k] (transposed)

  const int tid  = threadIdx.x;
  const int lane = tid & 31;
  const int wave = tid >> 5;
  const int wm = wave & 3;      // 0..3 : wave's 16-row slice
  const int wn = wave >> 2;     // 0..1 : wave's 32-col slice
  const int rowBase = blockIdx.x * 64;
  const int colBase = blockIdx.y * 64;

  const int half = lane >> 4;   // 0: lanes 0-15, 1: lanes 16-31
  const int l16  = lane & 15;

  v8f acc0 = {};
  v8f acc1 = {};

  // A tile loader mapping: 64 rows x 32 cols, 8 floats per thread.
  const int ar = tid >> 2;            // 0..63
  const int ak = (tid & 3) * 8;       // 0,8,16,24
  // B tile loader mapping: 32 (k) x 64 (n), 8 floats per thread, stored N-major.
  const int bk = tid >> 3;            // 0..31
  const int bn = (tid & 7) * 8;       // 0..56

  for (int k0 = 0; k0 < K; k0 += 32) {
    // ---- stage A (f32 -> f16) ----
    const float4* ap4 = (const float4*)(A + (size_t)(rowBase + ar) * K + k0 + ak);
    float4 a0 = ap4[0], a1 = ap4[1];
    v8h av = { (_Float16)a0.x, (_Float16)a0.y, (_Float16)a0.z, (_Float16)a0.w,
               (_Float16)a1.x, (_Float16)a1.y, (_Float16)a1.z, (_Float16)a1.w };
    *(v8h*)(As + ar * 40 + ak) = av;

    // ---- stage B transposed: Bs[n][k] ----
    const float4* bp4 = (const float4*)(B + (size_t)(k0 + bk) * N + colBase + bn);
    float4 b0 = bp4[0], b1 = bp4[1];
    Bs[(bn + 0) * 40 + bk] = (_Float16)b0.x;
    Bs[(bn + 1) * 40 + bk] = (_Float16)b0.y;
    Bs[(bn + 2) * 40 + bk] = (_Float16)b0.z;
    Bs[(bn + 3) * 40 + bk] = (_Float16)b0.w;
    Bs[(bn + 4) * 40 + bk] = (_Float16)b1.x;
    Bs[(bn + 5) * 40 + bk] = (_Float16)b1.y;
    Bs[(bn + 6) * 40 + bk] = (_Float16)b1.z;
    Bs[(bn + 7) * 40 + bk] = (_Float16)b1.w;

    // Pull next K-tile toward the WGP while this tile is consumed
    // (lowers to global_prefetch_b8).
    if (k0 + 32 < K) {
      __builtin_prefetch(A + (size_t)(rowBase + ar) * K + k0 + 32 + ak, 0, 1);
      __builtin_prefetch(B + (size_t)(k0 + 32 + bk) * N + colBase + bn, 0, 1);
    }
    __syncthreads();

    // ---- A fragment (contiguous k order, same as B) ----
    // lanes 0-15:  halves = K0..15 (chunks 0,1)
    // lanes 16-31: halves = K16..31 (chunks 2,3)
    const v8h* arow = (const v8h*)(As + (wm * 16 + l16) * 40);
    v8h alo = arow[half * 2];
    v8h ahi = arow[half * 2 + 1];
    v16h afrag = __builtin_shufflevector(alo, ahi,
        0,1,2,3,4,5,6,7,8,9,10,11,12,13,14,15);

    // ---- B fragments: lane holds column N, K 0..15 (lanes<16) / 16..31 ----
    const v8h* bcol0 = (const v8h*)(Bs + (wn * 32 + l16) * 40);
    v8h blo0 = bcol0[half * 2];
    v8h bhi0 = bcol0[half * 2 + 1];
    v16h bfrag0 = __builtin_shufflevector(blo0, bhi0,
        0,1,2,3,4,5,6,7,8,9,10,11,12,13,14,15);
    const v8h* bcol1 = (const v8h*)(Bs + (wn * 32 + 16 + l16) * 40);
    v8h blo1 = bcol1[half * 2];
    v8h bhi1 = bcol1[half * 2 + 1];
    v16h bfrag1 = __builtin_shufflevector(blo1, bhi1,
        0,1,2,3,4,5,6,7,8,9,10,11,12,13,14,15);

    acc0 = __builtin_amdgcn_wmma_f32_16x16x32_f16(
        false, afrag, false, bfrag0, (short)0, acc0, false, false);
    acc1 = __builtin_amdgcn_wmma_f32_16x16x32_f16(
        false, afrag, false, bfrag1, (short)0, acc1, false, false);
    __syncthreads();
  }

  // ---- epilogue (C/D layout: VGPR j -> M=j (lanes<16) / M=8+j) ----
  const int col0 = colBase + wn * 32 + l16;
  const int col1 = col0 + 16;
  const float bv0 = bias ? bias[col0] : 0.0f;
  const float bv1 = bias ? bias[col1] : 0.0f;
#pragma unroll
  for (int j = 0; j < 8; ++j) {
    int row = rowBase + wm * 16 + half * 8 + j;
    float v0 = acc0[j] + bv0;
    float v1 = acc1[j] + bv1;
    if (act == 1) {
      v0 = fmaxf(v0, 0.0f);
      v1 = fmaxf(v1, 0.0f);
    } else if (act == 2) {
      v0 = v0 / (1.0f + __expf(-v0));   // silu = x*sigmoid(x)
      v1 = v1 / (1.0f + __expf(-v1));
    }
    C[(size_t)row * N + col0] = v0;
    C[(size_t)row * N + col1] = v1;
  }
}

// ---------------------------------------------------------------------------
// Loan embedding: cont MLP (12->64 relu ->32), concat with 4 categorical
// embeddings (32+32+16+16+32 = 128), @ oW + ob.  One block (128 thr) per batch.
// ---------------------------------------------------------------------------
__global__ void loan_embed_kernel(
    const int* __restrict__ acI, const int* __restrict__ btI,
    const int* __restrict__ rtI, const int* __restrict__ atI,
    const float* __restrict__ cont,
    const float* __restrict__ ac, const float* __restrict__ bt,
    const float* __restrict__ rt, const float* __restrict__ at,
    const float* __restrict__ cW1, const float* __restrict__ cb1,
    const float* __restrict__ cW2, const float* __restrict__ cb2,
    const float* __restrict__ oW, const float* __restrict__ ob,
    float* __restrict__ loan) {
  __shared__ float comb[128];
  __shared__ float c1[64];
  const int b = blockIdx.x, tid = threadIdx.x;
  if (tid < 64) {
    float s = cb1[tid];
#pragma unroll
    for (int i = 0; i < 12; ++i) s += cont[b * 12 + i] * cW1[i * 64 + tid];
    c1[tid] = fmaxf(s, 0.0f);
  }
  __syncthreads();
  if (tid < 32) {
    float s = cb2[tid];
#pragma unroll
    for (int i = 0; i < 64; ++i) s += c1[i] * cW2[i * 32 + tid];
    comb[96 + tid] = s;
    comb[tid] = ac[acI[b] * 32 + tid];
  } else if (tid < 64) {
    comb[tid] = bt[btI[b] * 32 + (tid - 32)];
  } else if (tid < 80) {
    comb[tid] = rt[rtI[b] * 16 + (tid - 64)];
  } else if (tid < 96) {
    comb[tid] = at[atI[b] * 16 + (tid - 80)];
  }
  __syncthreads();
  float s = ob[tid];
#pragma unroll 8
  for (int j = 0; j < 128; ++j) s += comb[j] * oW[j * 128 + tid];
  loan[b * 128 + tid] = s;
}

// x = state_embed[state] + macro@mW + mb + loan_emb (bcast) + PE(s,d)
__global__ void build_x_kernel(
    const int* __restrict__ stateSeq, const float* __restrict__ macro,
    const float* __restrict__ stEmb, const float* __restrict__ mW,
    const float* __restrict__ mb, const float* __restrict__ loan,
    float* __restrict__ x) {
  const int row = blockIdx.x;          // b*S + s
  const int d = threadIdx.x;
  const int b = row / SEQ, s = row % SEQ;
  float m = mb[d];
#pragma unroll
  for (int i = 0; i < NMAC; ++i) m += macro[row * NMAC + i] * mW[i * 128 + d];
  const int st = stateSeq[row];
  const int i2 = d & ~1;               // 2*(d/2)
  float freq = __expf(-(float)i2 * (9.210340371976184f / 128.0f));
  float ang = (float)s * freq;
  float pe = (d & 1) ? __cosf(ang) : __sinf(ang);
  x[(size_t)row * 128 + d] = stEmb[st * 128 + d] + m + loan[b * 128 + d] + pe;
}

// ---------------------------------------------------------------------------
// Causal self-attention core, one block per (b,h).  HD=16, S=120.
// Flash-style single-pass online softmax; K/V tiles in LDS.
// ---------------------------------------------------------------------------
__global__ __launch_bounds__(128) void attn_kernel(
    const float* __restrict__ Q, const float* __restrict__ K,
    const float* __restrict__ V, float* __restrict__ O) {
  const int b = blockIdx.x / NHEAD, h = blockIdx.x % NHEAD;
  const int tid = threadIdx.x;
  __shared__ float ks[SEQ * HDIM];
  __shared__ float vs[SEQ * HDIM];
  for (int i = tid; i < SEQ * HDIM; i += 128) {
    int s = i >> 4, d = i & 15;
    size_t g = (size_t)(b * SEQ + s) * DIM + h * HDIM + d;
    ks[i] = K[g];
    vs[i] = V[g];
  }
  __syncthreads();
  if (tid >= SEQ) return;
  const int sq = tid;
  float q[HDIM], out[HDIM];
#pragma unroll
  for (int d = 0; d < HDIM; ++d) {
    q[d] = Q[(size_t)(b * SEQ + sq) * DIM + h * HDIM + d];
    out[d] = 0.0f;
  }
  float mrun = -3.0e38f, lrun = 0.0f;
  for (int s2 = 0; s2 <= sq; ++s2) {
    float sc = 0.0f;
#pragma unroll
    for (int d = 0; d < HDIM; ++d) sc += q[d] * ks[s2 * HDIM + d];
    sc *= 0.25f;                               // 1/sqrt(16)
    float mn = fmaxf(mrun, sc);
    float corr = __expf(mrun - mn);
    float p = __expf(sc - mn);
    lrun = lrun * corr + p;
#pragma unroll
    for (int d = 0; d < HDIM; ++d)
      out[d] = out[d] * corr + p * vs[s2 * HDIM + d];
    mrun = mn;
  }
  float inv = 1.0f / lrun;
#pragma unroll
  for (int d = 0; d < HDIM; ++d)
    O[(size_t)(b * SEQ + sq) * DIM + h * HDIM + d] = out[d] * inv;
}

// out = LN(x + y) ; if bcastY, y is indexed per-batch (row/S).  1 block / row.
__global__ __launch_bounds__(128) void add_ln_kernel(
    const float* __restrict__ x, const float* __restrict__ y,
    const float* __restrict__ g, const float* __restrict__ bta,
    float* __restrict__ out, int bcastY) {
  const int row = blockIdx.x, tid = threadIdx.x;
  const int lane = tid & 31, wv = tid >> 5;
  const size_t yrow = bcastY ? (size_t)(row / SEQ) : (size_t)row;
  float v = x[(size_t)row * 128 + tid] + y[yrow * 128 + tid];
  __shared__ float red[4];
  float s = v;
#pragma unroll
  for (int o = 16; o > 0; o >>= 1) s += __shfl_xor(s, o);
  if (lane == 0) red[wv] = s;
  __syncthreads();
  float mean = (red[0] + red[1] + red[2] + red[3]) * (1.0f / 128.0f);
  __syncthreads();
  float dd = v - mean;
  s = dd * dd;
#pragma unroll
  for (int o = 16; o > 0; o >>= 1) s += __shfl_xor(s, o);
  if (lane == 0) red[wv] = s;
  __syncthreads();
  float var = (red[0] + red[1] + red[2] + red[3]) * (1.0f / 128.0f);
  out[(size_t)row * 128 + tid] = dd * rsqrtf(var + 1e-5f) * g[tid] + bta[tid];
}

// state_logits[row,n] = x[row,:] @ headW[:,n] + headb[n]   (N=7: VALU)
__global__ void head_kernel(const float* __restrict__ x,
                            const float* __restrict__ W,
                            const float* __restrict__ bias,
                            float* __restrict__ out, int nrow) {
  int idx = blockIdx.x * blockDim.x + threadIdx.x;
  if (idx >= nrow * NST) return;
  int row = idx / NST, n = idx % NST;
  float s = bias[n];
#pragma unroll 8
  for (int k = 0; k < 128; ++k) s += x[(size_t)row * 128 + k] * W[k * NST + n];
  out[idx] = s;
}

// hazard[row] = sigmoid(h1[row,:64] @ hW2 + hb2) * 0.1
__global__ void hazard_kernel(const float* __restrict__ h1,
                              const float* __restrict__ W,
                              const float* __restrict__ bias,
                              float* __restrict__ out, int nrow) {
  int row = blockIdx.x * blockDim.x + threadIdx.x;
  if (row >= nrow) return;
  float s = bias[0];
#pragma unroll 8
  for (int j = 0; j < 64; ++j) s += h1[(size_t)row * 64 + j] * W[j];
  out[row] = 0.1f / (1.0f + __expf(-s));
}

// Deterministic hash -> standard normal (Box-Muller); JAX keys unreproducible
// without runtime, but this is stable across calls as the harness requires.
__device__ __forceinline__ unsigned hmix(unsigned x) {
  x ^= x >> 16; x *= 0x7feb352du; x ^= x >> 15; x *= 0x846ca68bu; x ^= x >> 16;
  return x;
}
__device__ __forceinline__ float hash_normal(unsigned a, unsigned b) {
  unsigned h1 = hmix(a * 0x9E3779B9u ^ b);
  unsigned h2 = hmix(h1 ^ (b * 0xC2B2AE35u + 0x85EBCA6Bu));
  float u1 = (float)((h1 >> 8) + 1u) * (1.0f / 16777217.0f);   // (0,1]
  float u2 = (float)(h2 >> 8) * (1.0f / 16777216.0f);
  return sqrtf(-2.0f * __logf(u1)) * __cosf(6.2831853071f * u2);
}

__global__ void diff_init_kernel(float* __restrict__ xd, int nrow) {
  int row = blockIdx.x * blockDim.x + threadIdx.x;
  if (row < nrow) xd[row] = hash_normal(0x13579BDFu, (unsigned)row);
}

// h1 = silu(ctx_proj + x*W1[128,:] + t_emb*W1[129,:] + b1)
__global__ void diff_step_a_kernel(
    const float* __restrict__ ctx, const float* __restrict__ xd,
    const float* __restrict__ W1, const float* __restrict__ b1,
    const float* __restrict__ TE, int t, float* __restrict__ h1) {
  int idx = blockIdx.x * blockDim.x + threadIdx.x;
  if (idx >= NROW * 128) return;
  int row = idx >> 7, d = idx & 127;
  float te = TE[t];
  float h = ctx[idx] + xd[row] * W1[128 * 128 + d] + te * W1[129 * 128 + d] + b1[d];
  h1[idx] = h / (1.0f + __expf(-h));
}

// pred = h2 @ W3 + b3 ; x = (x - c1*pred)*c2 + c3*noise
__global__ void diff_step_b_kernel(
    const float* __restrict__ h2, const float* __restrict__ W3,
    const float* __restrict__ b3, float* __restrict__ xd,
    float c1, float c2, float c3, int noiseIdx, int zeroNoise, int nrow) {
  int row = blockIdx.x * blockDim.x + threadIdx.x;
  if (row >= nrow) return;
  float s = b3[0];
#pragma unroll 8
  for (int j = 0; j < 128; ++j) s += h2[(size_t)row * 128 + j] * W3[j];
  float nz = zeroNoise ? 0.0f
                       : hash_normal(0x2468ACE0u + (unsigned)noiseIdx, (unsigned)row);
  xd[row] = (xd[row] - c1 * s) * c2 + c3 * nz;
}

__global__ void copy_kernel(const float* __restrict__ src,
                            float* __restrict__ dst, int n) {
  int i = blockIdx.x * blockDim.x + threadIdx.x;
  if (i < n) dst[i] = src[i];
}

// ---------------------------------------------------------------------------
// Host orchestration.  Input leaf order assumed = setup_inputs() insertion
// order with params pytree expanded in dict/list insertion order.
// ---------------------------------------------------------------------------
extern "C" void kernel_launch(void* const* d_in, const int* in_sizes, int n_in,
                              void* d_out, int out_size, void* d_ws, size_t ws_size,
                              hipStream_t stream) {
  (void)in_sizes; (void)n_in; (void)out_size; (void)ws_size;

  auto F = [&](int i) { return (const float*)d_in[i]; };
  auto I = [&](int i) { return (const int*)d_in[i]; };
  // layer l, leaf j (sa:0-7, ca:8-15, ln1:16-17, ln2:18-19, ln3:20-21,
  //                  ffW1:22 ffb1:23 ffW2:24 ffb2:25)
  auto LP = [&](int l, int j) { return (const float*)d_in[20 + l * 26 + j]; };

  float* outp = (float*)d_out;
  float* out_logits  = outp;                       // 30720*7
  float* out_hazard  = outp + NROW * NST;          // 30720
  float* out_pay     = outp + NROW * NST + NROW;   // 30720

  // Workspace carve-up.
  char* wsb = (char*)d_ws;
  size_t off = 0;
  auto wsalloc = [&](size_t bytes) {
    void* p = wsb + off;
    off = (off + bytes + 255) & ~(size_t)255;
    return p;
  };
  const size_t actB = (size_t)NROW * DIM * sizeof(float);
  float* X    = (float*)wsalloc(actB);                          // trunk state
  float* Qb   = (float*)wsalloc(actB);                          // q / proj / h1
  float* Kb   = (float*)wsalloc(actB);                          // k / h2
  float* Vb   = (float*)wsalloc(actB);                          // v
  float* ATT  = (float*)wsalloc(actB);                          // attn out / diff h1
  float* FF   = (float*)wsalloc((size_t)NROW * DFFN * sizeof(float)); // ffn / ctx
  float* LOAN = (float*)wsalloc((size_t)BCNT * DIM * sizeof(float));
  float* VMEM = (float*)wsalloc((size_t)BCNT * DIM * sizeof(float));
  float* CAB  = (float*)wsalloc((size_t)BCNT * DIM * sizeof(float));
  float* XD   = (float*)wsalloc((size_t)NROW * sizeof(float));

  auto gemm = [&](const float* A, const float* Bw, const float* bias, float* C,
                  int M, int N, int K, int act) {
    dim3 g(M / 64, N / 64);
    wmma_gemm_kernel<<<g, 256, 0, stream>>>(A, Bw, bias, C, M, N, K, act);
  };

  // ---- embeddings ----
  loan_embed_kernel<<<BCNT, 128, 0, stream>>>(
      I(0), I(1), I(2), I(3), F(4),
      F(7), F(8), F(9), F(10), F(11), F(12), F(13), F(14), F(15), F(16), LOAN);
  build_x_kernel<<<NROW, 128, 0, stream>>>(
      I(6), F(5), F(17), F(18), F(19), LOAN, X);

  // ---- transformer trunk ----
  for (int l = 0; l < NLAYER; ++l) {
    // self-attention
    gemm(X, LP(l, 0), LP(l, 4), Qb, NROW, DIM, DIM, 0);   // Wq,bq
    gemm(X, LP(l, 1), LP(l, 5), Kb, NROW, DIM, DIM, 0);   // Wk,bk
    gemm(X, LP(l, 2), LP(l, 6), Vb, NROW, DIM, DIM, 0);   // Wv,bv
    attn_kernel<<<BCNT * NHEAD, 128, 0, stream>>>(Qb, Kb, Vb, ATT);
    gemm(ATT, LP(l, 3), LP(l, 7), Qb, NROW, DIM, DIM, 0); // Wo,bo
    add_ln_kernel<<<NROW, 128, 0, stream>>>(X, Qb, LP(l, 16), LP(l, 17), X, 0);

    // cross-attention over a single memory token: softmax == 1, so
    // ca(x) = (loan_emb @ Wv + bv) @ Wo + bo, broadcast over S.
    gemm(LOAN, LP(l, 10), LP(l, 14), VMEM, BCNT, DIM, DIM, 0);  // ca.Wv,bv
    gemm(VMEM, LP(l, 11), LP(l, 15), CAB,  BCNT, DIM, DIM, 0);  // ca.Wo,bo
    add_ln_kernel<<<NROW, 128, 0, stream>>>(X, CAB, LP(l, 18), LP(l, 19), X, 1);

    // FFN
    gemm(X,  LP(l, 22), LP(l, 23), FF, NROW, DFFN, DIM, 1);  // relu
    gemm(FF, LP(l, 24), LP(l, 25), Qb, NROW, DIM, DFFN, 0);
    add_ln_kernel<<<NROW, 128, 0, stream>>>(X, Qb, LP(l, 20), LP(l, 21), X, 0);
  }

  // ---- heads ----
  head_kernel<<<(NROW * NST + 255) / 256, 256, 0, stream>>>(
      X, F(124), F(125), out_logits, NROW);
  gemm(X, F(126), F(127), Qb, NROW, 64, DIM, 1);             // hW1 relu
  hazard_kernel<<<(NROW + 255) / 256, 256, 0, stream>>>(
      Qb, F(128), F(129), out_hazard, NROW);

  // ---- diffusion sampler ----
  const float* TE = F(130);
  const float* W1 = F(131); const float* b1 = F(132);
  const float* W2 = F(133); const float* b2 = F(134);
  const float* W3 = F(135); const float* b3 = F(136);

  // context @ W1[:128] is step-invariant: one WMMA GEMM for all 100 steps.
  float* CTX = FF;   // trunk FFN buffer is free now
  gemm(X, W1, nullptr, CTX, NROW, DIM, DIM, 0);
  diff_init_kernel<<<(NROW + 255) / 256, 256, 0, stream>>>(XD, NROW);

  // host-side schedule
  double betas[TSTEP], alphas[TSTEP], acp[TSTEP];
  double run = 1.0;
  for (int t = 0; t < TSTEP; ++t) {
    betas[t] = 1e-4 + (0.02 - 1e-4) * (double)t / (double)(TSTEP - 1);
    alphas[t] = 1.0 - betas[t];
    run *= alphas[t];
    acp[t] = run;
  }

  for (int i = 0; i < TSTEP; ++i) {
    int t = TSTEP - 1 - i;
    float c1 = (float)(betas[t] / sqrt(1.0 - acp[t]));
    float c2 = (float)(1.0 / sqrt(alphas[t]));
    float c3 = (float)sqrt(betas[t]);
    diff_step_a_kernel<<<(NROW * 128 + 255) / 256, 256, 0, stream>>>(
        CTX, XD, W1, b1, TE, t, ATT);
    gemm(ATT, W2, b2, Kb, NROW, DIM, DIM, 2);                // silu fused
    diff_step_b_kernel<<<(NROW + 255) / 256, 256, 0, stream>>>(
        Kb, W3, b3, XD, c1, c2, c3, i, (t == 0) ? 1 : 0, NROW);
  }

  copy_kernel<<<(NROW + 255) / 256, 256, 0, stream>>>(XD, out_pay, NROW);
}
